// MtpDrafterANE_67242007987136
// MI455X (gfx1250) — matrix operand order repaired and checked by reading, not compiled, not run on hardware
//
#include <hip/hip_runtime.h>
#include <math.h>

// ---------------------------------------------------------------------------
// MI455X (gfx1250) decode-step kernel set.
// Batch-1 transformer decode: pure HBM-bandwidth problem (~820 MB fp32 read,
// ~35us floor at 23.3 TB/s). GEMVs use one wave32 per row with b128 loads;
// attention (the only real matrix-matrix shape: 4 heads sharing one KV head)
// uses v_wmma_f32_16x16x32_f16 with M padded 4->16.
// ---------------------------------------------------------------------------

typedef __attribute__((ext_vector_type(16))) _Float16 v16h;
typedef __attribute__((ext_vector_type(8)))  float    v8f;

#define EPS 1e-6f
#define NH 4

// ---------------- generic GEMV: out[n] = dot(x, W[n,:]), W row-major (N,K) --
__global__ void gemv_f32(const float* __restrict__ W, const float* __restrict__ x,
                         float* __restrict__ out, int N, int K) {
  int wave = (int)((blockIdx.x * blockDim.x + threadIdx.x) >> 5);
  int lane = threadIdx.x & 31;
  if (wave >= N) return;
  const float* row = W + (size_t)wave * K;
  float acc = 0.f;
  for (int k = lane * 4; k < K; k += 128) {
    float4 w4 = *(const float4*)(row + k);
    float4 x4 = *(const float4*)(x + k);
    acc += w4.x * x4.x + w4.y * x4.y + w4.z * x4.z + w4.w * x4.w;
  }
  #pragma unroll
  for (int off = 16; off; off >>= 1) acc += __shfl_down(acc, off, 32);
  if (lane == 0) out[wave] = acc;
}

// ---------------- RMSNorm: out = in * rsqrt(mean(in^2)+eps) * (1+w) ---------
__global__ void rms_kernel(const float* __restrict__ in, const float* __restrict__ w,
                           float* __restrict__ out, int n) {
  __shared__ float red[256];
  __shared__ float s_scale;
  float ss = 0.f;
  for (int i = threadIdx.x; i < n; i += 256) { float v = in[i]; ss += v * v; }
  red[threadIdx.x] = ss; __syncthreads();
  for (int s = 128; s; s >>= 1) {
    if ((int)threadIdx.x < s) red[threadIdx.x] += red[threadIdx.x + s];
    __syncthreads();
  }
  if (threadIdx.x == 0) s_scale = rsqrtf(red[0] / (float)n + EPS);
  __syncthreads();
  float sc = s_scale;
  for (int i = threadIdx.x; i < n; i += 256) {
    float v = in[i] * sc;
    out[i] = w ? v * (1.f + w[i]) : v;
  }
}

// ---------------- residual add of RMSNorm'd tensor: x += rms(t)*(1+w) -------
__global__ void add_rms_kernel(float* __restrict__ x, const float* __restrict__ t,
                               const float* __restrict__ w, int n) {
  __shared__ float red[256];
  __shared__ float s_scale;
  float ss = 0.f;
  for (int i = threadIdx.x; i < n; i += 256) { float v = t[i]; ss += v * v; }
  red[threadIdx.x] = ss; __syncthreads();
  for (int s = 128; s; s >>= 1) {
    if ((int)threadIdx.x < s) red[threadIdx.x] += red[threadIdx.x + s];
    __syncthreads();
  }
  if (threadIdx.x == 0) s_scale = rsqrtf(red[0] / (float)n + EPS);
  __syncthreads();
  float sc = s_scale;
  for (int i = threadIdx.x; i < n; i += 256)
    x[i] += t[i] * sc * (1.f + w[i]);
}

// ---------------- per-head q RMSNorm (+optional qnorm weight) + RoPE --------
__global__ void qprep_kernel(float* __restrict__ q, const float* __restrict__ qnorm_w,
                             const int* __restrict__ pos_ids, int hd, float theta) {
  __shared__ float buf[512];
  __shared__ float red[512];
  __shared__ float s_scale;
  int h = blockIdx.x, d = threadIdx.x;
  float v = 0.f;
  if (d < hd) { v = q[h * hd + d]; buf[d] = v; }
  red[threadIdx.x] = v * v; __syncthreads();
  for (int s = 256; s; s >>= 1) {
    if ((int)threadIdx.x < s) red[threadIdx.x] += red[threadIdx.x + s];
    __syncthreads();
  }
  if (threadIdx.x == 0) s_scale = rsqrtf(red[0] / (float)hd + EPS);
  __syncthreads();
  if (d < hd) {
    float sc = s_scale;
    int half = hd >> 1;
    int dm = (d < half) ? d : d - half;
    int dp = (d < half) ? d + half : d - half;
    float nv = v * sc;
    float pv = buf[dp] * sc;
    if (qnorm_w) { nv *= (1.f + qnorm_w[d]); pv *= (1.f + qnorm_w[dp]); }
    float ang = (float)pos_ids[0] * powf(theta, -2.f * (float)dm / (float)hd);
    float c = cosf(ang), s = sinf(ang);
    float outv = (d < half) ? (nv * c - pv * s) : (nv * c + pv * s);
    q[h * hd + d] = outv;
  }
}

// ---------------- attention scores via WMMA: S[h,t] = q[h,:] . K[t,:] -------
// One wave per 16-position tile. M dim = heads (4 used, 12 zero-padded).
__global__ void attn_scores_wmma(const float* __restrict__ Kc, const float* __restrict__ q,
                                 float* __restrict__ S, int ctx, int hd) {
  __shared__ float qs[NH * 512];
  int lane = threadIdx.x;
  for (int i = lane; i < NH * hd; i += 32) qs[i] = q[i];
  __syncthreads();
  int rowbase = blockIdx.x * 16;
  int n  = lane & 15;   // A: M index (head). B/C: N index (context col)
  int hi = lane >> 4;   // which K-half this lane holds
  v8f c = {};
  for (int kb = 0; kb < hd; kb += 32) {
    v16h a, b;
    int hh = n & 3;
    #pragma unroll
    for (int i = 0; i < 16; ++i) {
      // A 16x32: lane holds row n, k range hi*16..hi*16+15 (matches B)
      int koff = hi * 16 + i;
      float val = qs[hh * hd + kb + koff];
      a[i] = (_Float16)((n < NH) ? val : 0.f);
    }
    // B 32x16: lane holds column n, k range hi*16..hi*16+15 (contiguous mem)
    const float* krow = Kc + (size_t)(rowbase + n) * hd + kb + hi * 16;
    #pragma unroll
    for (int i = 0; i < 16; ++i) b[i] = (_Float16)krow[i];
    c = __builtin_amdgcn_wmma_f32_16x16x32_f16(false, a, false, b, (short)0, c,
                                               false, false);
  }
  // C layout: VGPR r, lanes 0-15 -> M=r, N=lane. Heads live in c[0..3].
  if (lane < 16) {
    #pragma unroll
    for (int h = 0; h < NH; ++h) S[(size_t)h * ctx + rowbase + lane] = c[h];
  }
}

// ---------------- softmax over context per head (mask added, scale = 1) -----
__global__ void softmax_kernel(float* __restrict__ S, const float* __restrict__ mask,
                               int ctx) {
  __shared__ float red[1024];
  for (int h = 0; h < NH; ++h) {
    float* s = S + (size_t)h * ctx;
    float m = -1e30f;
    for (int i = threadIdx.x; i < ctx; i += 1024) m = fmaxf(m, s[i] + mask[i]);
    red[threadIdx.x] = m; __syncthreads();
    for (int st = 512; st; st >>= 1) {
      if ((int)threadIdx.x < st)
        red[threadIdx.x] = fmaxf(red[threadIdx.x], red[threadIdx.x + st]);
      __syncthreads();
    }
    m = red[0]; __syncthreads();
    float sum = 0.f;
    for (int i = threadIdx.x; i < ctx; i += 1024) {
      float e = __expf(s[i] + mask[i] - m);
      s[i] = e; sum += e;
    }
    red[threadIdx.x] = sum; __syncthreads();
    for (int st = 512; st; st >>= 1) {
      if ((int)threadIdx.x < st) red[threadIdx.x] += red[threadIdx.x + st];
      __syncthreads();
    }
    float inv = 1.f / red[0]; __syncthreads();
    for (int i = threadIdx.x; i < ctx; i += 1024) s[i] *= inv;
  }
}

// ---------------- attn @ V via WMMA, split-K over context segments ----------
// P[seg][h*hd + d] = sum_{t in seg} attn[h,t] * V[t,d]
__global__ void av_wmma(const float* __restrict__ V, const float* __restrict__ attn,
                        float* __restrict__ P, int ctx, int hd, int seglen) {
  int lane = threadIdx.x;
  int dbase = blockIdx.x * 16;
  int seg = blockIdx.y;
  int n  = lane & 15;
  int hi = lane >> 4;
  v8f c = {};
  int tend = (seg + 1) * seglen;
  for (int tb = seg * seglen; tb < tend; tb += 32) {
    v16h a, b;
    int hh = n & 3;
    #pragma unroll
    for (int i = 0; i < 16; ++i) {
      int koff = hi * 16 + i;
      float val = attn[(size_t)hh * ctx + tb + koff];
      a[i] = (_Float16)((n < NH) ? val : 0.f);
    }
    #pragma unroll
    for (int i = 0; i < 16; ++i)
      b[i] = (_Float16)V[(size_t)(tb + hi * 16 + i) * hd + dbase + n];
    c = __builtin_amdgcn_wmma_f32_16x16x32_f16(false, a, false, b, (short)0, c,
                                               false, false);
  }
  if (lane < 16) {
    #pragma unroll
    for (int h = 0; h < NH; ++h)
      P[((size_t)seg * NH + h) * hd + dbase + lane] = c[h];
  }
}

__global__ void reduce_av(const float* __restrict__ P, float* __restrict__ a,
                          int nseg, int n) {
  for (int i = threadIdx.x; i < n; i += 256) {
    float s = 0.f;
    for (int g = 0; g < nseg; ++g) s += P[(size_t)g * n + i];
    a[i] = s;
  }
}

// ---------------- fused gate/up GEMV + gelu(tanh)*up ------------------------
__global__ void ffn_gate_up(const float* __restrict__ Gw, const float* __restrict__ Uw,
                            const float* __restrict__ h, float* __restrict__ f,
                            int N, int K) {
  int wave = (int)((blockIdx.x * blockDim.x + threadIdx.x) >> 5);
  int lane = threadIdx.x & 31;
  if (wave >= N) return;
  const float* gr = Gw + (size_t)wave * K;
  const float* ur = Uw + (size_t)wave * K;
  float ag = 0.f, au = 0.f;
  for (int k = lane * 4; k < K; k += 128) {
    float4 g4 = *(const float4*)(gr + k);
    float4 u4 = *(const float4*)(ur + k);
    float4 x4 = *(const float4*)(h + k);
    ag += g4.x * x4.x + g4.y * x4.y + g4.z * x4.z + g4.w * x4.w;
    au += u4.x * x4.x + u4.y * x4.y + u4.z * x4.z + u4.w * x4.w;
  }
  #pragma unroll
  for (int off = 16; off; off >>= 1) {
    ag += __shfl_down(ag, off, 32);
    au += __shfl_down(au, off, 32);
  }
  if (lane == 0) {
    float xg = ag;
    float g = 0.5f * xg * (1.f + tanhf(0.7978845608f * (xg + 0.044715f * xg * xg * xg)));
    f[wave] = g * au;
  }
}

// ---------------- lm_head GEMV + tanh softcap + per-block argmax ------------
__global__ void lm_head_kernel(const float* __restrict__ W, const float* __restrict__ h,
                               float* __restrict__ bmax, int* __restrict__ bidx,
                               int N, int K) {
  __shared__ float vmax[8];
  __shared__ int vrow[8];
  int wid = threadIdx.x >> 5, lane = threadIdx.x & 31;
  int row = blockIdx.x * 8 + wid;
  const float* r = W + (size_t)row * K;
  float acc = 0.f;
  for (int k = lane * 4; k < K; k += 128) {
    __builtin_prefetch(r + k + 2048, 0, 0);  // -> global_prefetch_b8
    float4 w4 = *(const float4*)(r + k);
    float4 x4 = *(const float4*)(h + k);
    acc += w4.x * x4.x + w4.y * x4.y + w4.z * x4.z + w4.w * x4.w;
  }
  #pragma unroll
  for (int off = 16; off; off >>= 1) acc += __shfl_down(acc, off, 32);
  if (lane == 0) { vmax[wid] = 30.f * tanhf(acc / 30.f); vrow[wid] = row; }
  __syncthreads();
  if (threadIdx.x == 0) {
    float m = vmax[0]; int mi = vrow[0];
    #pragma unroll
    for (int i = 1; i < 8; ++i)
      if (vmax[i] > m) { m = vmax[i]; mi = vrow[i]; }
    bmax[blockIdx.x] = m; bidx[blockIdx.x] = mi;
  }
}

__global__ void argmax_reduce(const float* __restrict__ bmax, const int* __restrict__ bidx,
                              int nb, float* __restrict__ out) {
  __shared__ float rm[256];
  __shared__ int ri[256];
  float m = -1e30f; int mi = 0x7fffffff;
  for (int i = threadIdx.x; i < nb; i += 256) {
    float v = bmax[i]; int id = bidx[i];
    if (v > m || (v == m && id < mi)) { m = v; mi = id; }
  }
  rm[threadIdx.x] = m; ri[threadIdx.x] = mi; __syncthreads();
  for (int s = 128; s; s >>= 1) {
    if ((int)threadIdx.x < s) {
      if (rm[threadIdx.x + s] > rm[threadIdx.x] ||
          (rm[threadIdx.x + s] == rm[threadIdx.x] && ri[threadIdx.x + s] < ri[threadIdx.x])) {
        rm[threadIdx.x] = rm[threadIdx.x + s];
        ri[threadIdx.x] = ri[threadIdx.x + s];
      }
    }
    __syncthreads();
  }
  if (threadIdx.x == 0) {
    ((int*)out)[0] = ri[0];   // token_id (int32 bit pattern)
    out[1] = rm[0];           // token_logit
  }
}

// ---------------------------------------------------------------------------
extern "C" void kernel_launch(void* const* d_in, const int* in_sizes, int n_in,
                              void* d_out, int out_size, void* d_ws, size_t ws_size,
                              hipStream_t stream) {
  const float* act          = (const float*)d_in[0];
  const int*   pos          = (const int*)d_in[1];
  const float* swa_mask     = (const float*)d_in[2];
  const float* full_mask    = (const float*)d_in[3];
  const float* kv13_k       = (const float*)d_in[4];
  const float* kv13_v       = (const float*)d_in[5];
  const float* kv14_k       = (const float*)d_in[6];
  const float* kv14_v       = (const float*)d_in[7];
  const float* pre_proj_w   = (const float*)d_in[8];
  const float* l_q_w        = (const float*)d_in[9];
  const float* l_o_w        = (const float*)d_in[10];
  const float* l_gate_w     = (const float*)d_in[11];
  const float* l_up_w       = (const float*)d_in[12];
  const float* l_down_w     = (const float*)d_in[13];
  const float* l_norms      = (const float*)d_in[14];
  const float* f_q_w        = (const float*)d_in[15];
  const float* f_o_w        = (const float*)d_in[16];
  const float* f_gate_w     = (const float*)d_in[17];
  const float* f_up_w       = (const float*)d_in[18];
  const float* f_down_w     = (const float*)d_in[19];
  const float* f_norms      = (const float*)d_in[20];
  const float* f_qnorm_w    = (const float*)d_in[21];
  const float* final_norm_w = (const float*)d_in[22];
  const float* lm_head_w    = (const float*)d_in[23];
  const float* post_proj_w  = (const float*)d_in[24];

  float* ws  = (float*)d_ws;
  float* x   = ws;             // 1024
  float* h   = x + 1024;       // 1024
  float* q   = h + 1024;       // 2048
  float* S   = q + 2048;       // 4*8192
  float* P   = S + 32768;      // 16*4*512
  float* a   = P + 32768;      // 2048
  float* f   = a + 2048;       // 4096
  float* t   = f + 4096;       // 4096
  float* bmax = t + 4096;      // 16384
  int*   bidx = (int*)(bmax + 16384);

  auto gemv = [&](const float* W, const float* xv, float* out, int N, int K) {
    gemv_f32<<<(N + 7) / 8, 256, 0, stream>>>(W, xv, out, N, K);
  };

  auto run_layer = [&](const float* qw, const float* ow, const float* gw,
                       const float* uw, const float* dw, const float* norms,
                       const float* qnw, const float* Kc, const float* Vc,
                       const float* mask, int hd, int ctx, float theta) {
    rms_kernel<<<1, 256, 0, stream>>>(x, norms, h, 1024);
    gemv(qw, h, q, NH * hd, 1024);
    qprep_kernel<<<NH, 512, 0, stream>>>(q, qnw, pos, hd, theta);
    attn_scores_wmma<<<ctx / 16, 32, 0, stream>>>(Kc, q, S, ctx, hd);
    softmax_kernel<<<1, 1024, 0, stream>>>(S, mask, ctx);
    int seglen = ctx / 16;
    av_wmma<<<dim3(hd / 16, 16), 32, 0, stream>>>(Vc, S, P, ctx, hd, seglen);
    reduce_av<<<1, 256, 0, stream>>>(P, a, 16, NH * hd);
    gemv(ow, a, t, 1024, NH * hd);
    add_rms_kernel<<<1, 256, 0, stream>>>(x, t, norms + 1024, 1024);
    rms_kernel<<<1, 256, 0, stream>>>(x, norms + 2048, h, 1024);
    ffn_gate_up<<<512, 256, 0, stream>>>(gw, uw, h, f, 4096, 1024);
    gemv(dw, f, t, 1024, 4096);
    add_rms_kernel<<<1, 256, 0, stream>>>(x, t, norms + 3072, 1024);
  };

  // pre-projection: (1,3072) @ (1024,3072)^T
  gemv(pre_proj_w, act, x, 1024, 3072);

  // 3 SWA layers (hd=256, ctx=2048, theta=1e4, no q-norm weight)
  for (int i = 0; i < 3; ++i) {
    run_layer(l_q_w + (size_t)i * 1024 * 1024,
              l_o_w + (size_t)i * 1024 * 1024,
              l_gate_w + (size_t)i * 4096 * 1024,
              l_up_w + (size_t)i * 4096 * 1024,
              l_down_w + (size_t)i * 1024 * 4096,
              l_norms + (size_t)i * 4096, nullptr,
              kv13_k, kv13_v, swa_mask, 256, 2048, 10000.f);
  }

  // full-attention layer (hd=512, ctx=8192, theta=1e6, with q-norm weight)
  run_layer(f_q_w, f_o_w, f_gate_w, f_up_w, f_down_w, f_norms, f_qnorm_w,
            kv14_k, kv14_v, full_mask, 512, 8192, 1000000.f);

  // final norm -> lm_head (softcap argmax/max) + post projection
  rms_kernel<<<1, 256, 0, stream>>>(x, final_norm_w, h, 1024);
  lm_head_kernel<<<131072 / 8, 256, 0, stream>>>(lm_head_w, h, bmax, bidx, 131072, 1024);
  argmax_reduce<<<1, 256, 0, stream>>>(bmax, bidx, 131072 / 8, (float*)d_out);
  gemv(post_proj_w, h, (float*)d_out + 2, 1536, 1024);
}
